// Mamba_encoder_10204842295805
// MI455X (gfx1250) — compile-verified, matrix operand order checked by reading, not run on hardware
//
#include <hip/hip_runtime.h>
#include <hip/hip_bf16.h>

typedef __attribute__((ext_vector_type(16))) _Float16 v16h;
typedef __attribute__((ext_vector_type(8)))  _Float16 v8h;
typedef __attribute__((ext_vector_type(8)))  float    v8f;

#define EMB      128
#define DSTATE   32
#define DCONV    8
#define DTRANK   8
#define WALK_LEN 64
#define N_WALKS  2048
#define USER_NUM 50000
#define NNODE    100000
#define XDBL_P   80   // 72 padded to 80 (5 N-tiles of 16)

__device__ __forceinline__ float silu_f(float x) { return x / (1.0f + __expf(-x)); }

__device__ __forceinline__ v8f wmma16(v16h a, v16h b, v8f c) {
  // D = A(16x32 f16) * B(32x16 f16) + C(16x16 f32)
  return __builtin_amdgcn_wmma_f32_16x16x32_f16(false, a, false, b, (short)0, c, false, false);
}

// A fragment: 16x32 f16 tile at (m0,k0) from row-major [*,stride] LDS array.
// wave32 layout: lanes 0-15 row M=lane, K {0..7,16..23}; lanes 16-31 same rows, K {8..15,24..31}.
// Two aligned 16B LDS loads per lane + register concat (no half-packing VALU).
__device__ __forceinline__ v16h frag_A(const _Float16* p, int stride, int m0, int k0) {
  const int lane = threadIdx.x & 31;
  const int hf = lane >> 4, r = lane & 15;
  const _Float16* q = p + (m0 + r) * stride + k0 + hf * 8;
  v8h lo = *(const v8h*)(q);
  v8h hi = *(const v8h*)(q + 16);
  return __builtin_shufflevector(lo, hi, 0, 1, 2, 3, 4, 5, 6, 7,
                                         8, 9, 10, 11, 12, 13, 14, 15);
}

// B fragment from pre-swizzled fragment-major weights: [tile][lane][16 halves].
// One aligned 32B vector load per lane (coalesced 1KB per wave per tile).
__device__ __forceinline__ v16h frag_Bp(const _Float16* __restrict__ packed, int tile) {
  const int lane = threadIdx.x & 31;
  return *(const v16h*)(packed + ((size_t)tile * 32 + lane) * 16);
}

// ---------------- helper kernels ----------------

__global__ void k_count(const int* __restrict__ idx, float* __restrict__ cnt, int n) {
  int i = blockIdx.x * 256 + threadIdx.x;
  if (i < n) atomicAdd(&cnt[idx[i]], 1.0f);
}

__global__ void k_invdenom(float* __restrict__ c, int n) {
  int i = blockIdx.x * 256 + threadIdx.x;
  if (i < n) c[i] = 1.0f / fmaxf(c[i], 1.0f);
}

// Pack f32 row-major [layers][128][ncols_src] weights into f16 fragment-major
// [layers][ntiles*4 tiles][32 lanes][16 halves], tile = nt*4 + kt.
// Element j of lane (hf,c) in tile (kt,nt) holds W[kt*32 + hf*8 + (j<8?j:8+j)][nt*16 + c].
__global__ void k_pack_B(const float* __restrict__ src, _Float16* __restrict__ dst,
                         int ncols_src, int ncols_valid, int ntiles, int total) {
  int i = blockIdx.x * 256 + threadIdx.x;
  if (i >= total) return;
  int j    = i & 15;
  int lane = (i >> 4) & 31;
  int tile = i >> 9;
  int tiles_per_layer = ntiles * 4;
  int layer = tile / tiles_per_layer;
  int t     = tile % tiles_per_layer;
  int nt = t >> 2, kt = t & 3;
  int hf = lane >> 4, c = lane & 15;
  int jj = (j < 8) ? j : (8 + j);
  int k  = kt * 32 + hf * 8 + jj;
  int n  = nt * 16 + c;
  float v = (n < ncols_valid) ? src[((size_t)layer * 128 + k) * ncols_src + n] : 0.0f;
  dst[i] = (_Float16)v;
}

// ---------------- main per-walk Mamba block ----------------
// 1 workgroup (8 wave32) per walk; everything LDS-resident.

__global__ void __launch_bounds__(256)
k_mamba_walk(const float* __restrict__ pu, const float* __restrict__ pi,
             const int* __restrict__ walk_node,
             const _Float16* __restrict__ Win_p,   // packed: 16*4 tiles
             const float* __restrict__ conv_w,     // [128][8]
             const float* __restrict__ conv_b,     // [128]
             const _Float16* __restrict__ Wx_p,    // packed: 5*4 tiles
             const float* __restrict__ W_dt,       // [8][128]
             const float* __restrict__ b_dt,       // [128]
             const float* __restrict__ A_log,      // [128][32]
             const float* __restrict__ D_skip,     // [128]
             const _Float16* __restrict__ Wout_p,  // packed: 8*4 tiles
             const float* __restrict__ inv_denom,  // [NNODE]
             float* __restrict__ s_out)            // [NNODE][128] (atomic accum)
{
  extern __shared__ char smem[];
  _Float16* x_h  = (_Float16*)(smem);                 // [64][128] 16KB  gathered input (f16)
  float*    xin  = (float*)   (smem + 16 * 1024);     // [64][128] 32KB  in-proj half 1
  float*    zbuf = (float*)   (smem + 48 * 1024);     // [64][128] 32KB  gate z
  float*    xc   = (float*)   (smem + 80 * 1024);     // [64][128] 32KB  conv+silu out (f32)
  _Float16* xc_h = (_Float16*)(smem + 112 * 1024);    // [64][128] 16KB  conv out (f16)
  float*    dbl  = (float*)   (smem + 128 * 1024);    // [64][80]  20KB  x-proj out (dt_low|B|C)
  float*    dtb  = (float*)   (smem + 148 * 1024);    // [64][128] 32KB  softplus(dt)
  float*    ybuf = (float*)   (smem + 180 * 1024);    // [64][128] 32KB  scan output
  _Float16* y_h  = (_Float16*)(smem + 212 * 1024);    // [64][128] 16KB  gated y (f16)
  int*      nidx = (int*)     (smem + 228 * 1024);    // [64]
  float*    ninv = (float*)   (smem + 228 * 1024 + 256); // [64]

  const int tid  = threadIdx.x;
  const int wave = tid >> 5;
  const int lane = tid & 31;
  const int hf   = lane >> 4;
  const int col  = lane & 15;
  const int walk = blockIdx.x;

  // ---- phase 0: node ids + 1/denom; prefetch this wave's weight tiles ----
  if (tid < WALK_LEN) {
    int node = walk_node[walk * WALK_LEN + tid];
    nidx[tid] = node;
    ninv[tid] = inv_denom[node];
  }
  // warm L2/WGP$ with packed weights (global_prefetch_b8)
  __builtin_prefetch(Win_p  + (size_t)wave * 8 * 512, 0, 3);
  __builtin_prefetch(Wout_p + (size_t)wave * 4 * 512, 0, 3);
  if (wave < 5) __builtin_prefetch(Wx_p + (size_t)wave * 4 * 512, 0, 3);
  __syncthreads();

  // ---- phase 1: gather ego -> x_h (f32 -> f16) ----
  {
    const int d = tid & 127;
    for (int t = tid >> 7; t < WALK_LEN; t += 2) {
      int node = nidx[t];
      float v = (node < USER_NUM) ? pu[(size_t)node * EMB + d]
                                  : pi[(size_t)(node - USER_NUM) * EMB + d];
      x_h[t * EMB + d] = (_Float16)v;
    }
  }
  __syncthreads();

  // ---- phase 2: GEMM1 in-proj  [64x128]@[128x256] -> xin | z ----
  for (int nt = wave; nt < 16; nt += 8) {
    const int n0 = nt * 16;
    v16h b0 = frag_Bp(Win_p, nt * 4 + 0);
    v16h b1 = frag_Bp(Win_p, nt * 4 + 1);
    v16h b2 = frag_Bp(Win_p, nt * 4 + 2);
    v16h b3 = frag_Bp(Win_p, nt * 4 + 3);
#pragma unroll
    for (int mt = 0; mt < 4; ++mt) {
      const int m0 = mt * 16;
      v8f acc = {};
      acc = wmma16(frag_A(x_h, EMB, m0, 0),  b0, acc);
      acc = wmma16(frag_A(x_h, EMB, m0, 32), b1, acc);
      acc = wmma16(frag_A(x_h, EMB, m0, 64), b2, acc);
      acc = wmma16(frag_A(x_h, EMB, m0, 96), b3, acc);
      float* dst = (n0 < EMB) ? (xin + n0) : (zbuf + (n0 - EMB));
#pragma unroll
      for (int v = 0; v < 8; ++v)
        dst[(m0 + v + hf * 8) * EMB + col] = acc[v];
    }
  }
  __syncthreads();

  // ---- phase 3: causal depthwise conv(8) + bias + SiLU ----
  {
    const int d = tid & 127;
    const float cb = conv_b[d];
    float cw[DCONV];
#pragma unroll
    for (int j = 0; j < DCONV; ++j) cw[j] = conv_w[d * DCONV + j];
    for (int t = tid >> 7; t < WALK_LEN; t += 2) {
      float s = cb;
#pragma unroll
      for (int j = 0; j < DCONV; ++j) {
        int tt = t - (DCONV - 1) + j;
        if (tt >= 0) s += xin[tt * EMB + d] * cw[j];
      }
      float v = silu_f(s);
      xc[t * EMB + d]   = v;
      xc_h[t * EMB + d] = (_Float16)v;
    }
  }
  __syncthreads();

  // ---- phase 4: GEMM2 x-proj  [64x128]@[128x80] -> dbl (dt_low|B|C, padded) ----
  for (int tile = wave; tile < 20; tile += 8) {
    const int m0 = (tile & 3) * 16;
    const int nt = tile >> 2;
    const int n0 = nt * 16;
    v8f acc = {};
#pragma unroll
    for (int kk = 0; kk < 4; ++kk)
      acc = wmma16(frag_A(xc_h, EMB, m0, kk * 32), frag_Bp(Wx_p, nt * 4 + kk), acc);
#pragma unroll
    for (int v = 0; v < 8; ++v)
      dbl[(m0 + v + hf * 8) * XDBL_P + n0 + col] = acc[v];
  }
  __syncthreads();

  // ---- phase 5: dt = softplus(dt_low @ W_dt + b_dt)   (K=8, VALU) ----
  {
    const int d = tid & 127;
    float wcol[DTRANK];
#pragma unroll
    for (int r = 0; r < DTRANK; ++r) wcol[r] = W_dt[r * EMB + d];
    const float bb = b_dt[d];
    for (int t = tid >> 7; t < WALK_LEN; t += 2) {
      float s = bb;
#pragma unroll
      for (int r = 0; r < DTRANK; ++r) s += dbl[t * XDBL_P + r] * wcol[r];
      dtb[t * EMB + d] = (s > 20.0f) ? s : __logf(1.0f + __expf(s));
    }
  }
  __syncthreads();

  // ---- phase 6: selective scan (sequential over t; 2 lanes per channel) ----
  {
    const int d  = tid >> 1;            // channel 0..127
    const int nb = (tid & 1) * 16;      // state half 0..15 / 16..31
    float A[16], h[16];
#pragma unroll
    for (int j = 0; j < 16; ++j) {
      A[j] = -__expf(A_log[d * DSTATE + nb + j]);
      h[j] = 0.0f;
    }
    for (int t = 0; t < WALK_LEN; ++t) {
      const float dtv = dtb[t * EMB + d];
      const float dtx = dtv * xc[t * EMB + d];
      const float* bc = dbl + t * XDBL_P;
      float acc = 0.0f;
#pragma unroll
      for (int j = 0; j < 16; ++j) {
        float dA = __expf(dtv * A[j]);
        h[j] = dA * h[j] + dtx * bc[DTRANK + nb + j];
        acc += h[j] * bc[DTRANK + DSTATE + nb + j];
      }
      acc += __shfl_xor(acc, 1, 32);    // combine the two state halves
      if ((tid & 1) == 0) ybuf[t * EMB + d] = acc;
    }
  }
  __syncthreads();

  // ---- phase 7: y = (y_scan + xc*D) * silu(z) -> f16 ----
  {
    const int d = tid & 127;
    const float Dk = D_skip[d];
    for (int t = tid >> 7; t < WALK_LEN; t += 2) {
      float yv = ybuf[t * EMB + d] + xc[t * EMB + d] * Dk;
      yv *= silu_f(zbuf[t * EMB + d]);
      y_h[t * EMB + d] = (_Float16)yv;
    }
  }
  __syncthreads();

  // ---- phase 8: GEMM3 out-proj [64x128]@[128x128] + fused scatter-mean ----
  {
    const int n0 = wave * 16;
    v16h b0 = frag_Bp(Wout_p, wave * 4 + 0);
    v16h b1 = frag_Bp(Wout_p, wave * 4 + 1);
    v16h b2 = frag_Bp(Wout_p, wave * 4 + 2);
    v16h b3 = frag_Bp(Wout_p, wave * 4 + 3);
#pragma unroll
    for (int mt = 0; mt < 4; ++mt) {
      const int m0 = mt * 16;
      v8f acc = {};
      acc = wmma16(frag_A(y_h, EMB, m0, 0),  b0, acc);
      acc = wmma16(frag_A(y_h, EMB, m0, 32), b1, acc);
      acc = wmma16(frag_A(y_h, EMB, m0, 64), b2, acc);
      acc = wmma16(frag_A(y_h, EMB, m0, 96), b3, acc);
#pragma unroll
      for (int v = 0; v < 8; ++v) {
        int t = m0 + v + hf * 8;
        float val = acc[v] * ninv[t];
        atomicAdd(&s_out[(size_t)nidx[t] * EMB + n0 + col], val);
      }
    }
  }
}

// ---------------- host-side orchestration ----------------

extern "C" void kernel_launch(void* const* d_in, const int* in_sizes, int n_in,
                              void* d_out, int out_size, void* d_ws, size_t ws_size,
                              hipStream_t stream) {
  (void)in_sizes; (void)n_in; (void)out_size; (void)ws_size;
  const float* user_emb = (const float*)d_in[0];
  const float* item_emb = (const float*)d_in[1];
  const int*   walk     = (const int*)  d_in[2];
  const float* W_in     = (const float*)d_in[3];   // [2][128][256]
  const float* conv_w   = (const float*)d_in[4];   // [2][128][8]
  const float* conv_b   = (const float*)d_in[5];   // [2][128]
  const float* W_x      = (const float*)d_in[6];   // [2][128][72]
  const float* W_dt     = (const float*)d_in[7];   // [2][8][128]
  const float* b_dt     = (const float*)d_in[8];   // [2][128]
  const float* A_log    = (const float*)d_in[9];   // [2][128][32]
  const float* D_skip   = (const float*)d_in[10];  // [2][128]
  const float* W_out    = (const float*)d_in[11];  // [2][128][128]
  float* out = (float*)d_out;                      // [100000][128]

  char* ws = (char*)d_ws;
  size_t off = 0;
  auto take = [&](size_t bytes) -> char* {
    char* p = ws + off;
    off = (off + bytes + 255) & ~(size_t)255;
    return p;
  };
  float*    cnt    = (float*)   take((size_t)NNODE * 4);        // counts -> inv_denom (in place)
  float*    ego1   = (float*)   take((size_t)NNODE * EMB * 4);  // layer-1 output
  // packed weights: [layer][ntiles*4][32][16] halves
  _Float16* Win_p  = (_Float16*)take((size_t)2 * 16 * 4 * 512 * 2);
  _Float16* Wx_p   = (_Float16*)take((size_t)2 * 5  * 4 * 512 * 2);
  _Float16* Wout_p = (_Float16*)take((size_t)2 * 8  * 4 * 512 * 2);

  const size_t smem = 228 * 1024 + 512;
  const int nWalkIdx = N_WALKS * WALK_LEN;

  // counts -> 1/denom
  hipMemsetAsync(cnt, 0, (size_t)NNODE * 4, stream);
  k_count<<<(nWalkIdx + 255) / 256, 256, 0, stream>>>(walk, cnt, nWalkIdx);
  k_invdenom<<<(NNODE + 255) / 256, 256, 0, stream>>>(cnt, NNODE);

  // fragment-major f16 weight packing (both layers)
  {
    int tot_in  = 2 * 16 * 4 * 512;
    int tot_x   = 2 * 5  * 4 * 512;
    int tot_out = 2 * 8  * 4 * 512;
    k_pack_B<<<(tot_in  + 255) / 256, 256, 0, stream>>>(W_in,  Win_p,  256, 256, 16, tot_in);
    k_pack_B<<<(tot_x   + 255) / 256, 256, 0, stream>>>(W_x,   Wx_p,    72,  72,  5, tot_x);
    k_pack_B<<<(tot_out + 255) / 256, 256, 0, stream>>>(W_out, Wout_p, 128, 128,  8, tot_out);
  }

  // layer 0: gather from user/item tables, accumulate into ego1
  hipMemsetAsync(ego1, 0, (size_t)NNODE * EMB * 4, stream);
  k_mamba_walk<<<N_WALKS, 256, smem, stream>>>(
      user_emb, item_emb, walk,
      Win_p, conv_w, conv_b, Wx_p, W_dt, b_dt, A_log, D_skip, Wout_p,
      cnt, ego1);

  // layer 1: gather from ego1, accumulate into d_out (= final ego)
  hipMemsetAsync(out, 0, (size_t)NNODE * EMB * 4, stream);
  k_mamba_walk<<<N_WALKS, 256, smem, stream>>>(
      ego1, ego1 + (size_t)USER_NUM * EMB, walk,
      Win_p + 16 * 4 * 512, conv_w + 128 * DCONV, conv_b + 128,
      Wx_p + 5 * 4 * 512, W_dt + DTRANK * 128, b_dt + 128,
      A_log + 128 * DSTATE, D_skip + 128, Wout_p + 8 * 4 * 512,
      cnt, out);
}
